// Dictionary_66090956750981
// MI455X (gfx1250) — compile-verified, matrix operand order checked by reading, not run on hardware
//
#include <hip/hip_runtime.h>

typedef unsigned int   u32;
typedef unsigned short u16;
typedef __bf16 bf16;

typedef __attribute__((ext_vector_type(16))) bf16  v16bf;
typedef __attribute__((ext_vector_type(8)))  float v8f;
typedef __attribute__((ext_vector_type(4)))  u32   v4u;
typedef __attribute__((ext_vector_type(4)))  float v4f;

#define DIM    512
#define NSLOT  4096
#define MTILE  16
#define NTILE  32
#define NWAVE  4
#define TPB    128

// ---- float -> bf16 (round to nearest even) ----
__device__ __forceinline__ u16 f2bf(float f) {
  u32 u = __float_as_uint(f);
  u32 r = u + 0x7FFFu + ((u >> 16) & 1u);
  return (u16)(r >> 16);
}
__device__ __forceinline__ u32 pack2bf(float a, float b) {
  return (u32)f2bf(a) | ((u32)f2bf(b) << 16);
}

// Load a 16x32 bf16 WMMA operand fragment: two contiguous 16B chunks per lane.
// (CDNA5 16-bit A/B layout: lane half selects the 8+8 K interleave; part0 ->
//  VGPR0-3 (frag elems 0..7), part1 -> VGPR4-7 (frag elems 8..15).)
__device__ __forceinline__ v16bf load_frag(const u16* p0, const u16* p1) {
  v16bf f;
  ((v4u*)&f)[0] = *(const v4u*)p0;
  ((v4u*)&f)[1] = *(const v4u*)p1;
  return f;
}

// ---------------- one-time weight preparation ----------------

// weight_c: plain f32 -> bf16 (row-major N x D already matches the B-fragment
// need: K (=D) contiguous per slot row).
__global__ void cvt_bf16_kernel(const float* __restrict__ src,
                                u16* __restrict__ dst, int n) {
  int i = blockIdx.x * 256 + threadIdx.x;
  if (i < n) dst[i] = f2bf(src[i]);
}

// weight_s: f32 -> bf16, pre-packed into exact B-fragment order so the hot
// loop reads two contiguous b128 per fragment.
// block = (slot/32)*32 + (d/16); within block:
//   idx = ((p*2 + h)*16 + c)*8 + j  with  k_local = p*16 + h*8 + j, c = d%16.
__global__ void pack_ws_kernel(const float* __restrict__ ws,
                               u16* __restrict__ dst, int n) {
  int i = blockIdx.x * 256 + threadIdx.x;
  if (i >= n) return;
  int slot = i >> 9;        // / DIM
  int d    = i & (DIM - 1);
  int skp  = slot >> 5;     // slot / 32
  int kl   = slot & 31;
  int ct   = d >> 4;        // d / 16
  int c    = d & 15;
  int p    = kl >> 4;
  int rem  = kl & 15;
  int h    = rem >> 3;
  int j    = rem & 7;
  int dest = (skp * 32 + ct) * 1024 + ((p * 2 + h) * 16 + c) * 8 + j;
  dst[dest] = f2bf(ws[i]);
}

// ---------------- fused flash-softmax dictionary kernel ----------------

__global__ __launch_bounds__(TPB, 1)
void dict_fused(const float* __restrict__ x,
                const u16*   __restrict__ wcb,   // weight_c bf16, row-major
                const u16*   __restrict__ wsb,   // weight_s bf16, frag-packed
                float*       __restrict__ out)
{
  __shared__ u16   xtile[MTILE * DIM];                 // 16 KB
  __shared__ float sred[2 * 2 * 8 * 32 * NWAVE];       // 16 KB (double buffered)
  __shared__ u16   pb[2 * MTILE * NTILE];              //  2 KB (double buffered)

  const int tid  = threadIdx.x;
  const int wave = tid >> 5;
  const int lane = tid & 31;
  const int l16  = lane & 15;
  const int hi   = (lane >> 4) & 1;
  const int tok0 = blockIdx.x * MTILE;
  const int dwave = wave * 128;

  // ---- stage x tile (16 x 512) as bf16 into LDS ----
  {
    const v4f* xf = (const v4f*)(x + (size_t)tok0 * DIM);
    u32* xt32 = (u32*)xtile;
#pragma unroll
    for (int i = 0; i < (MTILE * DIM / 4) / TPB; ++i) {  // 16 iters
      int e4 = i * TPB + tid;
      v4f q = xf[e4];
      xt32[e4 * 2 + 0] = pack2bf(q.x, q.y);
      xt32[e4 * 2 + 1] = pack2bf(q.z, q.w);
    }
  }
  __syncthreads();

  // ---- preload A fragments: this wave's 128-wide K slice of the x tile ----
  v16bf afrag[4];
#pragma unroll
  for (int c = 0; c < 4; ++c) {
    int k0 = wave * 128 + c * 32;
    const u16* p = &xtile[l16 * DIM + k0 + hi * 8];
    afrag[c] = load_frag(p, p + 16);
  }

  const v8f vzero = {0.f, 0.f, 0.f, 0.f, 0.f, 0.f, 0.f, 0.f};
  v8f oacc[8];
#pragma unroll
  for (int t = 0; t < 8; ++t) oacc[t] = vzero;

  float mrow[8], lrow[8];
#pragma unroll
  for (int v = 0; v < 8; ++v) { mrow[v] = -1e30f; lrow[v] = 0.0f; }

  for (int it = 0; it < NSLOT / NTILE; ++it) {   // 128 iterations
    const int nb  = it * NTILE;
    const int buf = it & 1;

    // ---- GEMM1: partial scores (16 x 32) over this wave's K slice ----
    v8f s[2];
    s[0] = vzero; s[1] = vzero;
#pragma unroll
    for (int sub = 0; sub < 2; ++sub) {
#pragma unroll
      for (int c = 0; c < 4; ++c) {
        int k0 = wave * 128 + c * 32;
        const u16* wp = wcb + (size_t)(nb + sub * 16 + l16) * DIM + k0 + hi * 8;
        v16bf b = load_frag(wp, wp + 16);
        s[sub] = __builtin_amdgcn_wmma_f32_16x16x32_bf16(
            false, afrag[c], false, b, (short)0, s[sub], false, false);
      }
    }

    // ---- reduce K-partials across the 4 waves through LDS ----
    float* sr = &sred[buf * 2048];
#pragma unroll
    for (int sub = 0; sub < 2; ++sub)
#pragma unroll
      for (int v = 0; v < 8; ++v)
        sr[((sub * 8 + v) * 32 + lane) * 4 + wave] = s[sub][v];
    __syncthreads();

    float sv[2][8];
#pragma unroll
    for (int sub = 0; sub < 2; ++sub)
#pragma unroll
      for (int v = 0; v < 8; ++v) {
        v4f q = *(const v4f*)&sr[((sub * 8 + v) * 32 + lane) * 4];
        sv[sub][v] = (q.x + q.y) + (q.z + q.w);
      }

    // ---- online softmax update (replicated per wave; identical results) ----
    u16* pbb = &pb[buf * (MTILE * NTILE)];
#pragma unroll
    for (int v = 0; v < 8; ++v) {
      float rm = fmaxf(sv[0][v], sv[1][v]);
#pragma unroll
      for (int msk = 1; msk <= 8; msk <<= 1)
        rm = fmaxf(rm, __shfl_xor(rm, msk, 32));
      float mn    = fmaxf(mrow[v], rm);
      float alpha = __expf(mrow[v] - mn);
      float p0    = __expf(sv[0][v] - mn);
      float p1    = __expf(sv[1][v] - mn);
      float rs    = p0 + p1;
#pragma unroll
      for (int msk = 1; msk <= 8; msk <<= 1)
        rs += __shfl_xor(rs, msk, 32);
      mrow[v] = mn;
      lrow[v] = lrow[v] * alpha + rs;
#pragma unroll
      for (int t = 0; t < 8; ++t) oacc[t][v] *= alpha;
      if (wave == 0) {                      // wave-uniform branch; no WMMA inside
        int row = v + hi * 8;
        pbb[row * NTILE + l16]      = f2bf(p0);
        pbb[row * NTILE + 16 + l16] = f2bf(p1);
      }
    }
    __syncthreads();

    // ---- GEMM2: out(16 x 128 slice) += P(16x32) @ Ws(32 x 128 slice) ----
    const u16* pp = &pbb[l16 * NTILE + hi * 8];
    v16bf pfrag = load_frag(pp, pp + 16);
#pragma unroll
    for (int t = 0; t < 8; ++t) {
      const u16* blockbase = wsb + (size_t)(it * 32 + wave * 8 + t) * 1024;
      const u16* q0 = blockbase + ((0 * 2 + hi) * 16 + l16) * 8;
      const u16* q1 = blockbase + ((1 * 2 + hi) * 16 + l16) * 8;
      v16bf b = load_frag(q0, q1);
      oacc[t] = __builtin_amdgcn_wmma_f32_16x16x32_bf16(
          false, pfrag, false, b, (short)0, oacc[t], false, false);
    }
  }

  // ---- epilogue: normalize by softmax denominator and store f32 ----
#pragma unroll
  for (int v = 0; v < 8; ++v) {
    float inv = 1.0f / lrow[v];
    int row = tok0 + v + hi * 8;
#pragma unroll
    for (int t = 0; t < 8; ++t)
      out[(size_t)row * DIM + dwave + t * 16 + l16] = oacc[t][v] * inv;
  }
}

// ---------------- host launcher ----------------

extern "C" void kernel_launch(void* const* d_in, const int* in_sizes, int n_in,
                              void* d_out, int out_size, void* d_ws, size_t ws_size,
                              hipStream_t stream) {
  const float* x  = (const float*)d_in[0];   // (8, 2048, 512) f32
  const float* ws = (const float*)d_in[1];   // weight_s (4096, 512) f32
  const float* wc = (const float*)d_in[2];   // weight_c (4096, 512) f32
  float* out = (float*)d_out;

  const int nW = NSLOT * DIM;                // 2M elements per weight
  u16* wcb = (u16*)d_ws;                     // [0, 4MB): weight_c bf16
  u16* wsb = wcb + nW;                       // [4MB, 8MB): weight_s bf16 packed

  cvt_bf16_kernel<<<(nW + 255) / 256, 256, 0, stream>>>(wc, wcb, nW);
  pack_ws_kernel<<<(nW + 255) / 256, 256, 0, stream>>>(ws, wsb, nW);

  const int ntok = in_sizes[0] / DIM;        // 16384
  dict_fused<<<ntok / MTILE, TPB, 0, stream>>>(x, wcb, wsb, out);
}
